// CNNLSTM_90520730730616
// MI455X (gfx1250) — matrix operand order repaired and checked by reading, not compiled, not run on hardware
//
#include <hip/hip_runtime.h>
#include <hip/hip_bf16.h>
#include <cstdint>

typedef __attribute__((ext_vector_type(16))) _Float16 v16h;
typedef __attribute__((ext_vector_type(8)))  _Float16 v8h;
typedef __attribute__((ext_vector_type(8)))  float    v8f;
typedef __attribute__((ext_vector_type(4)))  unsigned int v4u;
typedef __attribute__((ext_vector_type(8)))  int      v8i;
typedef __attribute__((ext_vector_type(4)))  int      v4i;

#define TT   32
#define NN   16
#define BB   (TT * NN)          // 512 samples
#define HID  512
#define FEAT 2048

#if __has_builtin(__builtin_amdgcn_tensor_load_to_lds) && __has_builtin(__builtin_amdgcn_s_wait_tensorcnt)
#define HAVE_TDM 1
#else
#define HAVE_TDM 0
#endif

// ---------------------------------------------------------------------------
// TDM: 1-D contiguous copy global -> LDS (D# per CDNA5 ISA).
// data_size = 8B units; nbytes must be a multiple of 8.
// ---------------------------------------------------------------------------
#if HAVE_TDM
__device__ __forceinline__ void tdm_load_1d(const void* gsrc, unsigned lds_byte_off,
                                            unsigned nbytes) {
    unsigned n8 = nbytes >> 3;                       // elements of 8 bytes
    unsigned long long ga = (unsigned long long)(uintptr_t)gsrc;
    v4u g0;
    g0.x = 1u;                                       // count=1, user descriptor
    g0.y = lds_byte_off;                             // lds_addr
    g0.z = (unsigned)ga;                             // global_addr[31:0]
    g0.w = ((unsigned)(ga >> 32) & 0x01FFFFFFu) | 0x80000000u;  // addr[56:32] | type=2
    v8i g1;
    g1[0] = (int)(3u << 16);                         // data_size = 3 (8B)
    g1[1] = (int)((n8 & 0xFFFFu) << 16);             // tensor_dim0[15:0]
    g1[2] = (int)(((n8 >> 16) & 0xFFFFu) | (1u << 16)); // tensor_dim0[31:16] | tensor_dim1=1
    g1[3] = (int)((n8 & 0xFFFFu) << 16);             // tile_dim0 = n8
    g1[4] = 1;                                       // tile_dim1 = 1, tile_dim2 = 0
    g1[5] = (int)n8;                                 // tensor_dim0_stride[31:0]
    g1[6] = 0;
    g1[7] = 0;
    v4i z4 = {0, 0, 0, 0};
#if defined(__clang_major__) && (__clang_major__ >= 23)
    v8i z8 = {0, 0, 0, 0, 0, 0, 0, 0};
    __builtin_amdgcn_tensor_load_to_lds(g0, g1, z4, z4, z8, 0);
#else
    __builtin_amdgcn_tensor_load_to_lds(g0, g1, z4, z4, 0);
#endif
}
#endif

// ---------------------------------------------------------------------------
// f32 -> f16 conversion
// ---------------------------------------------------------------------------
__global__ __launch_bounds__(256)
void cvt_f32_to_f16(const float* __restrict__ src, _Float16* __restrict__ dst, int n) {
    int i = blockIdx.x * blockDim.x + threadIdx.x;
    if (i < n) dst[i] = (_Float16)src[i];
}

// ---------------------------------------------------------------------------
// Conv weight repack. src shape (COUT, CIN, 2, 3, 3) f32.  off=(kd*3+ky)*3+kx
// ---------------------------------------------------------------------------
__global__ __launch_bounds__(256)
void repack_w_k(const float* __restrict__ src, _Float16* __restrict__ dst,
                int COUT, int CIN) {   // dst[co][off*CIN+ci]
    int i = blockIdx.x * blockDim.x + threadIdx.x;
    int total = COUT * CIN * 18;
    if (i >= total) return;
    int kx = i % 3; int t = i / 3;
    int ky = t % 3; t /= 3;
    int kd = t % 2; t /= 2;
    int ci = t % CIN;
    int co = t / CIN;
    int off = (kd * 3 + ky) * 3 + kx;
    dst[(size_t)co * (18 * CIN) + off * CIN + ci] = (_Float16)src[i];
}

__global__ __launch_bounds__(256)
void repack_w_cl(const float* __restrict__ src, _Float16* __restrict__ dst,
                 int COUT, int CIN) {  // dst[(off*COUT+co)*CIN+ci]
    int i = blockIdx.x * blockDim.x + threadIdx.x;
    int total = COUT * CIN * 18;
    if (i >= total) return;
    int kx = i % 3; int t = i / 3;
    int ky = t % 3; t /= 3;
    int kd = t % 2; t /= 2;
    int ci = t % CIN;
    int co = t / CIN;
    int off = (kd * 3 + ky) * 3 + kx;
    dst[((size_t)off * COUT + co) * CIN + ci] = (_Float16)src[i];
}

// ---------------------------------------------------------------------------
// Preprocess: x is already channels-last per sample; scale by 1/255 -> f16.
// ---------------------------------------------------------------------------
__global__ __launch_bounds__(256)
void preprocess_kernel(const float* __restrict__ x, _Float16* __restrict__ a0, int total) {
    int i = blockIdx.x * blockDim.x + threadIdx.x;
    if (i >= total) return;
    a0[i] = (_Float16)(x[i] * (1.0f / 255.0f));
}

// ---------------------------------------------------------------------------
// Layer-1 conv (CIN=4): scalar-gather implicit GEMM, channels-last in/out.
// ---------------------------------------------------------------------------
template<int CIN, int COUT, int DIN, int HIN, int WIN, int SHW>
__global__ __launch_bounds__(32)
void conv3d_gather_cl(const _Float16* __restrict__ act_in,
                      const _Float16* __restrict__ wK,       // [COUT][K]
                      const float*    __restrict__ bias,
                      _Float16*       __restrict__ act_out)  // [B][P][COUT]
{
    constexpr int DOUT = DIN - 1;
    constexpr int HOUT = HIN / SHW;
    constexpr int WOUT = WIN / SHW;
    constexpr int K    = CIN * 18;
    constexpr int P    = DOUT * HOUT * WOUT;

    const int lane  = threadIdx.x;
    const int half  = lane >> 4;
    const int c16   = lane & 15;
    const int b     = blockIdx.z;
    const int mbase = blockIdx.y * 16;
    const int pbase = blockIdx.x * 16;

    const int p   = pbase + c16;
    const int d_o = p / (HOUT * WOUT);
    const int rem = p % (HOUT * WOUT);
    const int y_o = rem / WOUT;
    const int x_o = rem % WOUT;

    const _Float16* __restrict__ arow = wK + (size_t)(mbase + c16) * K;
    const _Float16* __restrict__ ain  = act_in + (size_t)b * DIN * HIN * WIN * CIN;

    v8f acc = {};
    for (int k0 = 0; k0 < K; k0 += 32) {
        v16h A, B;
#pragma unroll
        for (int e = 0; e < 16; ++e) {
            int k = k0 + half * 16 + e;
            A[e] = (k < K) ? arow[k] : (_Float16)0.0f;
        }
#pragma unroll
        for (int e = 0; e < 16; ++e) {
            int k = k0 + half * 16 + e;
            _Float16 v = (_Float16)0.0f;
            if (k < K) {
                int ci  = k % CIN;
                int off = k / CIN;
                int kd  = off / 9;
                int r9  = off % 9;
                int ky  = r9 / 3;
                int kx  = r9 % 3;
                int zi  = d_o + kd;
                int yi  = y_o * SHW + ky - 1;
                int xi  = x_o * SHW + kx - 1;
                if (yi >= 0 && yi < HIN && xi >= 0 && xi < WIN)
                    v = ain[(((size_t)zi * HIN + yi) * WIN + xi) * CIN + ci];
            }
            B[e] = v;
        }
        acc = __builtin_amdgcn_wmma_f32_16x16x32_f16(false, A, false, B,
                                                     (short)0, acc, false, false);
    }
    v8h st;
#pragma unroll
    for (int r = 0; r < 8; ++r) {
        int co  = mbase + half * 8 + r;
        float v = acc[r] + bias[co];
        st[r] = (_Float16)(v > 0.0f ? v : 0.0f);
    }
    *(v8h*)(act_out + ((size_t)b * P + p) * COUT + mbase + half * 8) = st;
}

// ---------------------------------------------------------------------------
// Mid convs (L2..L5): 4 waves/block, 64 positions. Per-offset weight slice
// [COUT][CIN] staged in LDS (double-buffered), prefetched one offset ahead
// with the Tensor Data Mover (fallback: cooperative vector loads).
// ---------------------------------------------------------------------------
template<int CIN, int COUT, int DIN, int HIN, int WIN, int SHW>
__global__ __launch_bounds__(128)
void conv3d_cl_wmma_lds(const _Float16* __restrict__ act_in,
                        const _Float16* __restrict__ wrep,     // [18][COUT][CIN]
                        const float*    __restrict__ bias,
                        _Float16*       __restrict__ act_out)  // [B][P][COUT]
{
    constexpr int DOUT  = DIN - 1;
    constexpr int HOUT  = HIN / SHW;
    constexpr int WOUT  = WIN / SHW;
    constexpr int P     = DOUT * HOUT * WOUT;
    constexpr int MT    = COUT / 16;
    constexpr int SLICE = COUT * CIN;      // f16 elements per offset

    __shared__ _Float16 wlds[2][SLICE];

    const int tid   = threadIdx.x;
    const int wave  = tid >> 5;
    const int lane  = tid & 31;
    const int half  = lane >> 4;
    const int c16   = lane & 15;
    const int b     = blockIdx.z;
    const int pbase = (blockIdx.x * 4 + wave) * 16;

    const int p   = pbase + c16;
    const int d_o = p / (HOUT * WOUT);
    const int rem = p % (HOUT * WOUT);
    const int y_o = rem / WOUT;
    const int x_o = rem % WOUT;

    const _Float16* __restrict__ ain = act_in + (size_t)b * DIN * HIN * WIN * CIN;

    auto stage = [&](int off, int buf) {
        const _Float16* gsrc = wrep + (size_t)off * SLICE;
#if HAVE_TDM
        if (wave == 0)
            tdm_load_1d(gsrc, (unsigned)(uintptr_t)&wlds[buf][0], SLICE * 2);
#else
        for (int i = tid * 8; i < SLICE; i += 128 * 8)
            *(v8h*)&wlds[buf][i] = *(const v8h*)(gsrc + i);
#endif
    };

    v8f acc[MT];
#pragma unroll
    for (int mt = 0; mt < MT; ++mt) acc[mt] = v8f{};

    stage(0, 0);
#if HAVE_TDM
    if (wave == 0) __builtin_amdgcn_s_wait_tensorcnt(0);
#endif
    __syncthreads();

    for (int off = 0; off < 18; ++off) {
        const int cur = off & 1;
        if (off + 1 < 18) stage(off + 1, 1 - cur);

        const int kd = off / 9;
        const int r9 = off % 9;
        const int ky = r9 / 3;
        const int kx = r9 % 3;
        const int zi = d_o + kd;
        const int yi = y_o * SHW + ky - 1;
        const int xi = x_o * SHW + kx - 1;
        const bool valid = (yi >= 0) && (yi < HIN) && (xi >= 0) && (xi < WIN);
        const _Float16* __restrict__ bpos =
            ain + (((size_t)zi * HIN + yi) * WIN + xi) * CIN;
        const _Float16* __restrict__ wof = &wlds[cur][0];

        for (int ci0 = 0; ci0 < CIN; ci0 += 32) {
            v16h B = {};
            if (valid) B = *(const v16h*)(bpos + ci0 + half * 16);
#pragma unroll
            for (int mt = 0; mt < MT; ++mt) {
                const _Float16* wrow = wof + (size_t)(mt * 16 + c16) * CIN + ci0;
                v8h lo = *(const v8h*)(wrow + half * 16);
                v8h hi = *(const v8h*)(wrow + 8 + half * 16);
                v16h A;
#pragma unroll
                for (int e = 0; e < 8; ++e) { A[e] = lo[e]; A[e + 8] = hi[e]; }
                acc[mt] = __builtin_amdgcn_wmma_f32_16x16x32_f16(
                    false, A, false, B, (short)0, acc[mt], false, false);
            }
        }
#if HAVE_TDM
        if (off + 1 < 18 && wave == 0) __builtin_amdgcn_s_wait_tensorcnt(0);
#endif
        __syncthreads();
    }

#pragma unroll
    for (int mt = 0; mt < MT; ++mt) {
        v8h st;
#pragma unroll
        for (int r = 0; r < 8; ++r) {
            int co  = mt * 16 + half * 8 + r;
            float v = acc[mt][r] + bias[co];
            st[r] = (_Float16)(v > 0.0f ? v : 0.0f);
        }
        *(v8h*)(act_out + ((size_t)b * P + p) * COUT + mt * 16 + half * 8) = st;
    }
}

// ---------------------------------------------------------------------------
// L6: single wave per block (P = 16), weights from global.
// ---------------------------------------------------------------------------
template<int CIN, int COUT, int DIN, int HIN, int WIN, int SHW>
__global__ __launch_bounds__(32)
void conv3d_cl_wmma(const _Float16* __restrict__ act_in,
                    const _Float16* __restrict__ wrep,     // [18][COUT][CIN]
                    const float*    __restrict__ bias,
                    _Float16*       __restrict__ act_out)  // [B][P][COUT]
{
    constexpr int DOUT = DIN - 1;
    constexpr int HOUT = HIN / SHW;
    constexpr int WOUT = WIN / SHW;
    constexpr int P    = DOUT * HOUT * WOUT;
    constexpr int MT   = COUT / 16;

    const int lane  = threadIdx.x;
    const int half  = lane >> 4;
    const int c16   = lane & 15;
    const int b     = blockIdx.z;
    const int pbase = blockIdx.x * 16;

    const int p   = pbase + c16;
    const int d_o = p / (HOUT * WOUT);
    const int rem = p % (HOUT * WOUT);
    const int y_o = rem / WOUT;
    const int x_o = rem % WOUT;

    const _Float16* __restrict__ ain = act_in + (size_t)b * DIN * HIN * WIN * CIN;

    v8f acc[MT];
#pragma unroll
    for (int mt = 0; mt < MT; ++mt) acc[mt] = v8f{};

    for (int off = 0; off < 18; ++off) {
        const int kd = off / 9;
        const int r9 = off % 9;
        const int ky = r9 / 3;
        const int kx = r9 % 3;
        const int zi = d_o + kd;
        const int yi = y_o * SHW + ky - 1;
        const int xi = x_o * SHW + kx - 1;
        const bool valid = (yi >= 0) && (yi < HIN) && (xi >= 0) && (xi < WIN);
        const _Float16* __restrict__ bpos =
            ain + (((size_t)zi * HIN + yi) * WIN + xi) * CIN;
        const _Float16* __restrict__ wof = wrep + (size_t)off * COUT * CIN;

        for (int ci0 = 0; ci0 < CIN; ci0 += 32) {
            v16h B = {};
            if (valid) B = *(const v16h*)(bpos + ci0 + half * 16);
#pragma unroll
            for (int mt = 0; mt < MT; ++mt) {
                const _Float16* wrow = wof + (size_t)(mt * 16 + c16) * CIN + ci0;
                v8h lo = *(const v8h*)(wrow + half * 16);
                v8h hi = *(const v8h*)(wrow + 8 + half * 16);
                v16h A;
#pragma unroll
                for (int e = 0; e < 8; ++e) { A[e] = lo[e]; A[e + 8] = hi[e]; }
                acc[mt] = __builtin_amdgcn_wmma_f32_16x16x32_f16(
                    false, A, false, B, (short)0, acc[mt], false, false);
            }
        }
    }
#pragma unroll
    for (int mt = 0; mt < MT; ++mt) {
        v8h st;
#pragma unroll
        for (int r = 0; r < 8; ++r) {
            int co  = mt * 16 + half * 8 + r;
            float v = acc[mt][r] + bias[co];
            st[r] = (_Float16)(v > 0.0f ? v : 0.0f);
        }
        *(v8h*)(act_out + ((size_t)b * P + p) * COUT + mt * 16 + half * 8) = st;
    }
}

// ---------------------------------------------------------------------------
// feat transpose: [b][p=16][c=128] channels-last -> feat_cm[b][c*16 + p]
// ---------------------------------------------------------------------------
__global__ __launch_bounds__(256)
void feat_transpose(const _Float16* __restrict__ src, _Float16* __restrict__ dst) {
    int i = blockIdx.x * blockDim.x + threadIdx.x;
    if (i >= BB * FEAT) return;
    int c = i & 127;
    int t = i >> 7;
    int p = t & 15;
    int b = t >> 4;
    dst[(size_t)b * FEAT + c * 16 + p] = src[i];
}

// ---------------------------------------------------------------------------
// gates_x = feat @ W_ih^T + (b_ih + b_hh)   -- fully parallel, f32 out.
// One wave computes a 16x64 tile (A reused across 4 B tiles). K = 2048.
// ---------------------------------------------------------------------------
__global__ __launch_bounds__(32)
void gates_x_wmma(const _Float16* __restrict__ feat,   // [512][FEAT]
                  const _Float16* __restrict__ Wih16,  // [4*HID][FEAT]
                  const float*    __restrict__ b_ih,
                  const float*    __restrict__ b_hh,
                  float*          __restrict__ gx)     // [512][4*HID]
{
    constexpr int NT = 4;
    const int lane = threadIdx.x;
    const int half = lane >> 4;
    const int c16  = lane & 15;
    const int mb   = blockIdx.y * 16;
    const int nb0  = blockIdx.x * (16 * NT);

    const _Float16* __restrict__ xa = feat + (size_t)(mb + c16) * FEAT;

    v8f acc[NT];
#pragma unroll
    for (int nt = 0; nt < NT; ++nt) acc[nt] = v8f{};

    for (int k0 = 0; k0 < FEAT; k0 += 32) {
        v8h lo = *(const v8h*)(xa + k0 + half * 16);
        v8h hi = *(const v8h*)(xa + k0 + 8 + half * 16);
        v16h A;
#pragma unroll
        for (int e = 0; e < 8; ++e) { A[e] = lo[e]; A[e + 8] = hi[e]; }
#pragma unroll
        for (int nt = 0; nt < NT; ++nt) {
            v16h B = *(const v16h*)(Wih16 + (size_t)(nb0 + nt * 16 + c16) * FEAT
                                    + k0 + half * 16);
            acc[nt] = __builtin_amdgcn_wmma_f32_16x16x32_f16(
                false, A, false, B, (short)0, acc[nt], false, false);
        }
    }
#pragma unroll
    for (int nt = 0; nt < NT; ++nt) {
        const int n = nb0 + nt * 16 + c16;
        const float bb = b_ih[n] + b_hh[n];
#pragma unroll
        for (int r = 0; r < 8; ++r) {
            const int m = mb + r + half * 8;
            gx[(size_t)m * (4 * HID) + n] = acc[nt][r] + bb;
        }
    }
}

// ---------------------------------------------------------------------------
// LSTM recurrence: only h @ W_hh^T is serialized (K = 512). Accumulator is
// seeded from precomputed gates_x. Single workgroup, 16 waves; h,c in LDS.
// ---------------------------------------------------------------------------
__global__ __launch_bounds__(512)
void lstm_wmma_kernel(const float*    __restrict__ gx,     // [T*N][4*HID] f32
                      const float*    __restrict__ done,   // [T][N]
                      const float*    __restrict__ h0,
                      const float*    __restrict__ c0,
                      const _Float16* __restrict__ Whh16,  // [4*HID][HID]
                      _Float16*       __restrict__ hs,     // [T][N][HID] f16 out
                      float*          __restrict__ hT_out,
                      float*          __restrict__ cT_out)
{
    __shared__ float sh[NN * HID];   // 32 KB
    __shared__ float sc[NN * HID];   // 32 KB

    const int tid  = threadIdx.x;
    const int wave = tid >> 5;
    const int lane = tid & 31;
    const int half = lane >> 4;
    const int c16  = lane & 15;

    for (int i = tid; i < NN * HID; i += 512) { sh[i] = h0[i]; sc[i] = c0[i]; }
    __syncthreads();

    for (int t = 0; t < TT; ++t) {
        for (int i = tid; i < NN * HID; i += 512) {
            float keep = 1.0f - done[t * NN + (i >> 9)];
            sh[i] *= keep;
            sc[i] *= keep;
        }
        __syncthreads();

        v8f acc[8];
#pragma unroll
        for (int gt = 0; gt < 8; ++gt) {
            const int g  = gt >> 1;
            const int s  = gt & 1;
            const int nb = g * HID + wave * 32 + s * 16;

            // seed from gates_x
            v8f a;
#pragma unroll
            for (int r = 0; r < 8; ++r)
                a[r] = gx[(size_t)(t * NN + r + half * 8) * (4 * HID) + nb + c16];

            // h @ W_hh^T : K = HID
            for (int k0 = 0; k0 < HID; k0 += 32) {
                v16h A, B;
#pragma unroll
                for (int e = 0; e < 16; ++e)
                    A[e] = (_Float16)sh[c16 * HID + k0 + half * 16 + e];
                B = *(const v16h*)(Whh16 + (size_t)(nb + c16) * HID + k0 + half * 16);
                a = __builtin_amdgcn_wmma_f32_16x16x32_f16(false, A, false, B,
                                                           (short)0, a, false, false);
            }
            acc[gt] = a;
        }
        __syncthreads();

#pragma unroll
        for (int s = 0; s < 2; ++s) {
            const int n = wave * 32 + s * 16 + c16;
#pragma unroll
            for (int r = 0; r < 8; ++r) {
                const int m = r + half * 8;
                float gi = acc[0 + s][r];
                float gf = acc[2 + s][r];
                float gg = acc[4 + s][r];
                float go = acc[6 + s][r];
                float si = 1.0f / (1.0f + __expf(-gi));
                float sf = 1.0f / (1.0f + __expf(-gf));
                float so = 1.0f / (1.0f + __expf(-go));
                float tg = tanhf(gg);
                float cc = sf * sc[m * HID + n] + si * tg;
                float hh = so * tanhf(cc);
                sc[m * HID + n] = cc;
                sh[m * HID + n] = hh;
                hs[((size_t)t * NN + m) * HID + n] = (_Float16)hh;
            }
        }
        __syncthreads();
    }

    for (int i = tid; i < NN * HID; i += 512) { hT_out[i] = sh[i]; cT_out[i] = sc[i]; }
}

// ---------------------------------------------------------------------------
// FC: out[512,512] = relu(hs @ Wf^T + bf)
// ---------------------------------------------------------------------------
__global__ __launch_bounds__(32)
void fc_wmma_kernel(const _Float16* __restrict__ hsf,
                    const _Float16* __restrict__ Wf16,
                    const float*    __restrict__ bf,
                    float*          __restrict__ out)
{
    const int lane = threadIdx.x;
    const int half = lane >> 4;
    const int c16  = lane & 15;
    const int mb   = blockIdx.y * 16;
    const int nb   = blockIdx.x * 16;

    const _Float16* __restrict__ xa = hsf + (size_t)(mb + c16) * HID;

    v8f acc = {};
    for (int k0 = 0; k0 < HID; k0 += 32) {
        v8h lo = *(const v8h*)(xa + k0 + half * 16);
        v8h hi = *(const v8h*)(xa + k0 + 8 + half * 16);
        v16h A;
#pragma unroll
        for (int e = 0; e < 8; ++e) { A[e] = lo[e]; A[e + 8] = hi[e]; }
        v16h B = *(const v16h*)(Wf16 + (size_t)(nb + c16) * HID + k0 + half * 16);
        acc = __builtin_amdgcn_wmma_f32_16x16x32_f16(false, A, false, B,
                                                     (short)0, acc, false, false);
    }
#pragma unroll
    for (int r = 0; r < 8; ++r) {
        const int m = mb + r + half * 8;
        const int n = nb + c16;
        float v = acc[r] + bf[n];
        out[(size_t)m * HID + n] = v > 0.0f ? v : 0.0f;
    }
}

// ---------------------------------------------------------------------------
// Host launcher
// ---------------------------------------------------------------------------
static inline size_t align_up(size_t v, size_t a) { return (v + a - 1) & ~(a - 1); }

extern "C" void kernel_launch(void* const* d_in, const int* in_sizes, int n_in,
                              void* d_out, int out_size, void* d_ws, size_t ws_size,
                              hipStream_t stream) {
    (void)in_sizes; (void)n_in; (void)out_size; (void)ws_size;

    const float* x    = (const float*)d_in[0];
    const float* done = (const float*)d_in[1];
    const float* h0   = (const float*)d_in[2];
    const float* c0   = (const float*)d_in[3];
    const float* w11  = (const float*)d_in[4];  const float* b11 = (const float*)d_in[5];
    const float* w12  = (const float*)d_in[6];  const float* b12 = (const float*)d_in[7];
    const float* w21  = (const float*)d_in[8];  const float* b21 = (const float*)d_in[9];
    const float* w22  = (const float*)d_in[10]; const float* b22 = (const float*)d_in[11];
    const float* w31  = (const float*)d_in[12]; const float* b31 = (const float*)d_in[13];
    const float* w32  = (const float*)d_in[14]; const float* b32 = (const float*)d_in[15];
    const float* Wih  = (const float*)d_in[16];
    const float* Whh  = (const float*)d_in[17];
    const float* bih  = (const float*)d_in[18];
    const float* bhh  = (const float*)d_in[19];
    const float* Wf   = (const float*)d_in[20];
    const float* bf   = (const float*)d_in[21];

    float* out_fh = (float*)d_out;
    float* out_hT = out_fh + 512 * 512;
    float* out_cT = out_hT + 16 * 512;

    char*  base = (char*)d_ws;
    size_t cur  = 0;
    auto carveB = [&](size_t nbytes) -> void* {
        cur = align_up(cur, 256);
        void* p = (void*)(base + cur);
        cur += nbytes;
        return p;
    };
    auto carve16 = [&](size_t n) -> _Float16* { return (_Float16*)carveB(n * 2); };
    auto carve32 = [&](size_t n) -> float*    { return (float*)carveB(n * 4); };

    _Float16* R0 = carve16(20971520);   // a0 / a2 / a4 / feat_cl
    _Float16* R1 = carve16(100663296);  // a1 / a3 / a5
    _Float16* w11h = carve16(2304);     // [32][72]
    _Float16* w12h = carve16(18432);    // [18][32][32]
    _Float16* w21h = carve16(36864);    // [18][64][32]
    _Float16* w22h = carve16(73728);    // [18][64][64]
    _Float16* w31h = carve16(147456);   // [18][128][64]
    _Float16* w32h = carve16(294912);   // [18][128][128]
    _Float16* Wihh = carve16(4194304);
    _Float16* Whhh = carve16(1048576);
    _Float16* Wfh  = carve16(262144);
    _Float16* fcm  = carve16(1048576);  // feat channel-major [B][2048]
    _Float16* hsb  = carve16(262144);   // [T][N][HID]
    float*    gxb  = carve32(BB * 4 * HID);  // [512][2048] f32 gate preactivations

    auto rpK = [&](const float* s, _Float16* d, int co, int ci) {
        int n = co * ci * 18;
        repack_w_k<<<(n + 255) / 256, 256, 0, stream>>>(s, d, co, ci);
    };
    auto rpCL = [&](const float* s, _Float16* d, int co, int ci) {
        int n = co * ci * 18;
        repack_w_cl<<<(n + 255) / 256, 256, 0, stream>>>(s, d, co, ci);
    };
    auto cvt = [&](const float* s, _Float16* d, int n) {
        cvt_f32_to_f16<<<(n + 255) / 256, 256, 0, stream>>>(s, d, n);
    };
    rpK (w11, w11h,  32,   4);
    rpCL(w12, w12h,  32,  32);
    rpCL(w21, w21h,  64,  32);
    rpCL(w22, w22h,  64,  64);
    rpCL(w31, w31h, 128,  64);
    rpCL(w32, w32h, 128, 128);
    cvt(Wih, Wihh, 4194304);
    cvt(Whh, Whhh, 1048576);
    cvt(Wf,  Wfh,  262144);

    {
        int total = BB * 7 * 32 * 32 * 4;
        preprocess_kernel<<<(total + 255) / 256, 256, 0, stream>>>(x, R0, total);
    }

    // conv stack (channels-last ping-pong R0 <-> R1)
    conv3d_gather_cl <  4,  32, 7, 32, 32, 1><<<dim3(384, 2, BB),  32, 0, stream>>>(R0, w11h, b11, R1);
    conv3d_cl_wmma_lds< 32,  32, 6, 32, 32, 2><<<dim3( 20, 1, BB), 128, 0, stream>>>(R1, w12h, b12, R0);
    conv3d_cl_wmma_lds< 32,  64, 5, 16, 16, 1><<<dim3( 16, 1, BB), 128, 0, stream>>>(R0, w21h, b21, R1);
    conv3d_cl_wmma_lds< 64,  64, 4, 16, 16, 2><<<dim3(  3, 1, BB), 128, 0, stream>>>(R1, w22h, b22, R0);
    conv3d_cl_wmma_lds< 64, 128, 3,  8,  8, 1><<<dim3(  2, 1, BB), 128, 0, stream>>>(R0, w31h, b31, R1);
    conv3d_cl_wmma    <128, 128, 2,  8,  8, 2><<<dim3(  1, 1, BB),  32, 0, stream>>>(R1, w32h, b32, R0);
    // R0 holds feat channels-last [B][16][128]

    {
        int total = BB * FEAT;
        feat_transpose<<<(total + 255) / 256, 256, 0, stream>>>(R0, fcm);
    }

    // Parallel input projections for all timesteps (bulk of LSTM FLOPs)
    gates_x_wmma<<<dim3((4 * HID) / 64, BB / 16), 32, 0, stream>>>(fcm, Wihh, bih, bhh, gxb);

    // Serialized recurrence (K=512 only) on one workgroup
    lstm_wmma_kernel<<<1, 512, 0, stream>>>(gxb, done, h0, c0, Whhh, hsb, out_hT, out_cT);

    fc_wmma_kernel<<<dim3(32, 32), 32, 0, stream>>>(hsb, Wfh, bf, out_fh);
}
